// Encoder_25031069401686
// MI455X (gfx1250) — matrix operand order repaired and checked by reading, not compiled, not run on hardware
//
#include <hip/hip_runtime.h>

#define FD 128  // feature dim (in_hidden == out_hidden)

typedef __attribute__((ext_vector_type(16))) __bf16 v16bf;
typedef __attribute__((ext_vector_type(8)))  float  v8f;

// ---------------------------------------------------------------- zero fill
__global__ void zero_f32(float* __restrict__ p, int n) {
    int i = blockIdx.x * blockDim.x + threadIdx.x;
    if (i < n) p[i] = 0.0f;
}

// ---------------------------------------------------------------- degrees
__global__ void degree_kernel(const int* __restrict__ src, const int* __restrict__ dst,
                              float* __restrict__ deg_out, float* __restrict__ deg_in, int E) {
    int i = blockIdx.x * blockDim.x + threadIdx.x;
    if (i < E) {
        unsafeAtomicAdd(&deg_out[src[i]], 1.0f);
        unsafeAtomicAdd(&deg_in[dst[i]],  1.0f);
    }
}

// ---------------------------------------------------------------- norms
__global__ void norm_kernel(const float* __restrict__ deg_out, const float* __restrict__ deg_in,
                            float* __restrict__ norm_src, float* __restrict__ norm_dst, int N) {
    int i = blockIdx.x * blockDim.x + threadIdx.x;
    if (i < N) {
        norm_src[i] = rsqrtf(fmaxf(deg_out[i], 1.0f));
        norm_dst[i] = rsqrtf(fmaxf(deg_in[i],  1.0f));
    }
}

// ---------------------------------------------------------------- SpMM scatter
// agg[dst[e]] += feat[src[e]] * norm_src[src[e]]   (norm fused into gather)
// 32 lanes per edge (edge index is wave-uniform -> scalar loads), float4 per lane,
// hw f32 atomics into L2-resident accumulator.
__global__ void __launch_bounds__(256)
scatter_kernel(const float* __restrict__ feat, const int* __restrict__ src,
               const int* __restrict__ dst, const float* __restrict__ norm_src,
               float* __restrict__ agg, int E) {
    int e    = blockIdx.x * 8 + (threadIdx.x >> 5);
    int lane = threadIdx.x & 31;
    if (e >= E) return;
    int s = src[e], d = dst[e];
    float ns = norm_src[s];
    const float4* frow = (const float4*)(feat + (size_t)s * FD);
    float4 v = frow[lane];
    float* drow = agg + (size_t)d * FD + lane * 4;
    unsafeAtomicAdd(drow + 0, v.x * ns);
    unsafeAtomicAdd(drow + 1, v.y * ns);
    unsafeAtomicAdd(drow + 2, v.z * ns);
    unsafeAtomicAdd(drow + 3, v.w * ns);
}

// ---------------------------------------------------------------- W pre-pack (fragment-ready bf16)
// wpack layout: [ct (8)][kb (4)][lane (32)][16 bf16] -> 32 KB, per documented 16-bit
// B-matrix 32x16 layout: lane holds col (lane&15), K = kb*32 + 16*(p>=4) + 8*(lane>=16) + 2*(p&3) + i.
__global__ void wpack_kernel(const float* __restrict__ W, __bf16* __restrict__ wpack) {
    int idx  = blockIdx.x * blockDim.x + threadIdx.x;   // 0..1023
    int lane = idx & 31;
    int kb   = (idx >> 5) & 3;
    int ct   = idx >> 7;
    int half = lane >> 4;
    int c    = ct * 16 + (lane & 15);
    v16bf tmp;
#pragma unroll
    for (int p = 0; p < 8; ++p) {
        int k = kb * 32 + ((p >= 4) ? 16 : 0) + 8 * half + 2 * (p & 3);
        tmp[2 * p]     = (__bf16)W[(size_t)k * FD + c];
        tmp[2 * p + 1] = (__bf16)W[(size_t)(k + 1) * FD + c];
    }
    ((v16bf*)wpack)[idx] = tmp;
}

// ---------------------------------------------------------------- A pre-pack (fragment-ready bf16)
// apack layout: [rowTile][kb (4)][lane (32)][16 bf16], per documented 16-bit
// A-matrix 16x32 layout: lane holds row (lane&15),
// K = kb*32 + 16*(p>=4) + 8*(lane>=16) + 2*(p&3) + i.
__global__ void apack_kernel(const float* __restrict__ agg, __bf16* __restrict__ apack,
                             int N, int total) {
    int idx = blockIdx.x * blockDim.x + threadIdx.x;    // tiles*4*32 threads
    if (idx >= total) return;
    int lane = idx & 31;
    int kb   = (idx >> 5) & 3;
    int t    = idx >> 7;
    int half = lane >> 4;
    int row  = t * 16 + (lane & 15);
    const bool valid = (row < N);
    const float* arow = agg + (size_t)(valid ? row : 0) * FD;
    v16bf tmp;
#pragma unroll
    for (int p = 0; p < 8; ++p) {
        int k = kb * 32 + ((p >= 4) ? 16 : 0) + 8 * half + 2 * (p & 3);
        float f0 = valid ? arow[k]     : 0.0f;
        float f1 = valid ? arow[k + 1] : 0.0f;
        tmp[2 * p]     = (__bf16)f0;
        tmp[2 * p + 1] = (__bf16)f1;
    }
    ((v16bf*)apack)[idx] = tmp;
}

// ---------------------------------------------------------------- WMMA GEMM + fused epilogue
// h = prelu((agg @ W) * norm_dst[:,None] + b, a1), plus per-column sum/sumsq for BN.
// Block = 8 waves; wave w owns row tile (blk*8 + w): 16 rows x 128 cols.
// Pre-packed operands -> inner loop is pure b128 loads + v_wmma. BN partials are
// reduced through LDS (ds_add_f32) and flushed with 256 global atomics per block.
__global__ void __launch_bounds__(256)
gemm_kernel(const __bf16* __restrict__ apack, const __bf16* __restrict__ wpack,
            const float* __restrict__ bvec, const float* __restrict__ norm_dst,
            const float* __restrict__ a1, float* __restrict__ h,
            float* __restrict__ colsum, float* __restrict__ colsumsq,
            int N, int tiles) {
    __shared__ float s_sum[FD];
    __shared__ float s_sq[FD];

    const int tid  = threadIdx.x;
    const int lane = tid & 31;
    const int wave = tid >> 5;
    const int t    = blockIdx.x * 8 + wave;         // row tile (wave-uniform)
    const bool active = (t < tiles);

    if (tid < FD) { s_sum[tid] = 0.0f; s_sq[tid] = 0.0f; }
    __syncthreads();

    if (active) {                                   // wave-uniform: EXEC stays all-ones
        const int half    = lane >> 4;
        const int rowBase = t * 16;
        const int n0      = lane & 15;
        const v16bf* __restrict__ ap = (const v16bf*)apack;
        const v16bf* __restrict__ wp = (const v16bf*)wpack;

        // A fragments: 4 contiguous 32B loads per lane
        v16bf afrag[4];
#pragma unroll
        for (int kb = 0; kb < 4; ++kb)
            afrag[kb] = ap[(size_t)(t * 4 + kb) * 32 + lane];

        // hoisted per-row dst-norms (row set invariant across column tiles)
        const bool fullTile = (rowBase + 16 <= N);  // wave-uniform
        float nd[8];
#pragma unroll
        for (int v = 0; v < 8; ++v) {
            int r = rowBase + v + 8 * half;
            nd[v] = norm_dst[fullTile ? r : min(r, N - 1)];
        }
        const float a1v = a1[0];

        for (int ct = 0; ct < 8; ++ct) {
            const int c = ct * 16 + n0;             // this lane's output column
            v8f acc = {};
#pragma unroll
            for (int kb = 0; kb < 4; ++kb) {
                v16bf bfrag = wp[(size_t)(ct * 4 + kb) * 32 + lane];
                acc = __builtin_amdgcn_wmma_f32_16x16x32_bf16(
                          false, afrag[kb], false, bfrag, (short)0, acc, false, false);
            }
            // epilogue: C/D layout -> element v is row rowBase + v + 8*half, col c
            const float bias = bvec[c];
            float lsum = 0.0f, lsq = 0.0f;
            if (fullTile) {                         // branchless fast path
#pragma unroll
                for (int v = 0; v < 8; ++v) {
                    int r = rowBase + v + 8 * half;
                    float hv = acc[v] * nd[v] + bias;
                    hv = (hv >= 0.0f) ? hv : a1v * hv;     // PReLU (a1)
                    h[(size_t)r * FD + c] = hv;
                    lsum += hv;
                    lsq  += hv * hv;
                }
            } else {                                // guarded tail tile
#pragma unroll
                for (int v = 0; v < 8; ++v) {
                    int r = rowBase + v + 8 * half;
                    if (r < N) {
                        float hv = acc[v] * nd[v] + bias;
                        hv = (hv >= 0.0f) ? hv : a1v * hv;
                        h[(size_t)r * FD + c] = hv;
                        lsum += hv;
                        lsq  += hv * hv;
                    }
                }
            }
            atomicAdd(&s_sum[c], lsum);             // ds_add_f32
            atomicAdd(&s_sq[c],  lsq);
        }
    }
    __syncthreads();
    if (tid < FD) {
        unsafeAtomicAdd(&colsum[tid],   s_sum[tid]);
        unsafeAtomicAdd(&colsumsq[tid], s_sq[tid]);
    }
}

// ---------------------------------------------------------------- BN finalize (1 block, 128 thr)
__global__ void bn_finalize(const float* __restrict__ colsum, const float* __restrict__ colsumsq,
                            const float* __restrict__ gamma, const float* __restrict__ beta,
                            float* __restrict__ scale, float* __restrict__ shift, int N) {
    int c = threadIdx.x;
    float invN = 1.0f / (float)N;
    float mean = colsum[c] * invN;
    float var  = colsumsq[c] * invN - mean * mean;   // biased variance
    float s    = gamma[c] * rsqrtf(var + 1e-5f);
    scale[c] = s;
    shift[c] = beta[c] - mean * s;
}

// ---------------------------------------------------------------- BN apply + outer PReLU (in place)
__global__ void bn_apply(float* __restrict__ out, const float* __restrict__ scale,
                         const float* __restrict__ shift, const float* __restrict__ a2,
                         int total4) {
    int i = blockIdx.x * blockDim.x + threadIdx.x;
    if (i >= total4) return;
    int c4 = (i & 31) * 4;              // column group within the 128-wide row
    float4* p = (float4*)out + i;
    float4 v = *p;
    float a = a2[0];
    float x;
    x = v.x * scale[c4 + 0] + shift[c4 + 0]; v.x = (x >= 0.0f) ? x : a * x;
    x = v.y * scale[c4 + 1] + shift[c4 + 1]; v.y = (x >= 0.0f) ? x : a * x;
    x = v.z * scale[c4 + 2] + shift[c4 + 2]; v.z = (x >= 0.0f) ? x : a * x;
    x = v.w * scale[c4 + 3] + shift[c4 + 3]; v.w = (x >= 0.0f) ? x : a * x;
    *p = v;
}

// ---------------------------------------------------------------- launch
extern "C" void kernel_launch(void* const* d_in, const int* in_sizes, int n_in,
                              void* d_out, int out_size, void* d_ws, size_t ws_size,
                              hipStream_t stream) {
    const float* feat  = (const float*)d_in[0];
    const int*   src   = (const int*)  d_in[1];
    const int*   dst   = (const int*)  d_in[2];
    const float* W     = (const float*)d_in[3];
    const float* bvec  = (const float*)d_in[4];
    const float* a1    = (const float*)d_in[5];
    const float* gamma = (const float*)d_in[6];
    const float* beta  = (const float*)d_in[7];
    const float* a2    = (const float*)d_in[8];
    float* out = (float*)d_out;

    const int ND    = in_sizes[0];
    const int N     = ND / FD;
    const int E     = in_sizes[1];
    const int tiles = (N + 15) / 16;

    // workspace layout (floats); zeroed prefix first, packs at the end
    float* ws       = (float*)d_ws;
    float* agg      = ws;                       // N*128
    float* deg_out  = agg + (size_t)ND;         // N
    float* deg_in   = deg_out + N;              // N
    float* colsum   = deg_in + N;               // 128
    float* colsumsq = colsum + FD;              // 128
    float* norm_src = colsumsq + FD;            // N
    float* norm_dst = norm_src + N;             // N
    float* bn_scale = norm_dst + N;             // 128
    float* bn_shift = bn_scale + FD;            // 128
    __bf16* apack   = (__bf16*)(bn_shift + FD); // tiles*4*32*16 bf16
    __bf16* wpack   = apack + (size_t)tiles * 2048; // 8*4*32*16 bf16 = 32 KB

    // 1) zero agg + degrees + column stats (everything accumulated into)
    int zc = ND + 2 * N + 2 * FD;
    zero_f32<<<(zc + 255) / 256, 256, 0, stream>>>(ws, zc);

    // 2) degrees
    degree_kernel<<<(E + 255) / 256, 256, 0, stream>>>(src, dst, deg_out, deg_in, E);

    // 3) norms
    norm_kernel<<<(N + 255) / 256, 256, 0, stream>>>(deg_out, deg_in, norm_src, norm_dst, N);

    // 4) fused-scale gather/scatter SpMM (8 edges per 256-thread block)
    scatter_kernel<<<(E + 7) / 8, 256, 0, stream>>>(feat, src, dst, norm_src, agg, E);

    // 5) pre-pack W and A into WMMA fragment order (bf16)
    wpack_kernel<<<4, 256, 0, stream>>>(W, wpack);
    int atotal = tiles * 128;
    apack_kernel<<<(atotal + 255) / 256, 256, 0, stream>>>(agg, apack, N, atotal);

    // 6) WMMA GEMM + norm_dst/bias/PReLU epilogue + BN partial sums (h -> d_out)
    gemm_kernel<<<(tiles + 7) / 8, 256, 0, stream>>>(apack, wpack, bvec, norm_dst, a1,
                                                     out, colsum, colsumsq, N, tiles);

    // 7) BN scale/shift
    bn_finalize<<<1, FD, 0, stream>>>(colsum, colsumsq, gamma, beta, bn_scale, bn_shift, N);

    // 8) BN apply + outer PReLU, in place on d_out
    int total4 = N * (FD / 4);
    bn_apply<<<(total4 + 255) / 256, 256, 0, stream>>>(out, bn_scale, bn_shift, a2, total4);
}
